// SparseSDFVQVAE_10711648436603
// MI455X (gfx1250) — matrix operand order repaired and checked by reading, compiled, not run on hardware
//
#include <hip/hip_runtime.h>
#include <hip/hip_bf16.h>

typedef __attribute__((ext_vector_type(16))) __bf16 v16bf;
typedef __attribute__((ext_vector_type(8)))  float  v8f;
typedef __attribute__((ext_vector_type(4)))  unsigned int v4u;
typedef __attribute__((ext_vector_type(8)))  int  v8i;
typedef __attribute__((ext_vector_type(4)))  int  v4i;

#define N_PTS   65536
#define DIM     64
#define KCODES  8192
#define NBATCH  8
#define CTILES  (KCODES / 16)      // 512 column tiles
#define CHUNK   16                 // tiles per TDM chunk (64 KB)
#define NCHUNK  (CTILES / CHUNK)   // 32 chunks
#define CHUNK_BYTES (CHUNK * 4096) // 65536
#define TILE_U16    2048           // uint16 elems per 4KB tile block
#define BUF_U16     (CHUNK * TILE_U16)

// ---- workspace layout (bytes) ----
#define OFF_CSQ   0u                            // 8192*4 = 32768
#define OFF_B     32768u                        // 512 tiles * 4KB = 2 MiB
#define OFF_CNT   (32768u + 2097152u)           // 8*8192*4 = 262144
#define OFF_LOSS  (32768u + 2097152u + 262144u) // 4 bytes

// ---- output layout (float elements) ----
#define OUT_VQ    4194304u
#define OUT_COMM  4194305u
#define OUT_IDX   4194306u
#define OUT_PERP  4259842u
#define OUT_ENT   4259843u
#define OUT_UNIQ  4259844u

__device__ __forceinline__ unsigned short f2bf(float x) {
    unsigned int u = __float_as_uint(x);
    unsigned int r = u + 0x7FFFu + ((u >> 16) & 1u);
    return (unsigned short)(r >> 16);
}
__device__ __forceinline__ float bf2f(unsigned short h) {
    return __uint_as_float(((unsigned int)h) << 16);
}

// ---------------------------------------------------------------------------
__global__ void k_csq(const float* __restrict__ cb, float* __restrict__ csq) {
    int i = blockIdx.x * blockDim.x + threadIdx.x;
    if (i < KCODES) {
        const float* r = cb + (size_t)i * DIM;
        float s = 0.f;
#pragma unroll
        for (int j = 0; j < DIM; ++j) s = fmaf(r[j], r[j], s);
        csq[i] = s;
    }
}

// ---------------------------------------------------------------------------
// Pre-swizzle codebook into per-tile 4KB blocks, B-fragment lane layout.
// Tile block (uint16 elems): [hi h=0 |hi h=1 |lo h=0 |lo h=1], 512 each.
// Within a fragment: elem[lane*16 + e], where for B 32x16:
//   k_local = 16*(lane>>4) + e, n = tile*16 + (lane&15), k = 32*h + k_local.
// ---------------------------------------------------------------------------
__global__ void k_prepB(const float* __restrict__ cb, unsigned short* __restrict__ B) {
    int gid  = blockIdx.x * blockDim.x + threadIdx.x;
    int lane = gid & 31;
    int unit = gid >> 5;            // (tile, part)
    int part = unit & 3;            // 0:hi/h0 1:hi/h1 2:lo/h0 3:lo/h1
    int tile = unit >> 2;
    if (tile >= CTILES) return;
    int h  = part & 1, lo = part >> 1;
    int n  = tile * 16 + (lane & 15);
    int k0 = 32 * h + 16 * (lane >> 4);
    const float* src = cb + (size_t)n * DIM + k0;   // 16 consecutive floats
    unsigned short* dst = B + (size_t)tile * TILE_U16 + part * 512 + lane * 16;
#pragma unroll
    for (int e = 0; e < 16; ++e) {
        float x = src[e];
        unsigned short hi = f2bf(x);
        dst[e] = lo ? f2bf(x - bf2f(hi)) : hi;
    }
}

// ---------------------------------------------------------------------------
// TDM: DMA one contiguous 64KB chunk of B (16 tile blocks) into LDS.
// 1-D copy expressed as 2-D tensor: data_size=8B, dim0=tile0=8192 units.
// (6-arg builtin form: g0, g1, g2, g3, extra group, cpol)
// ---------------------------------------------------------------------------
__device__ __forceinline__ void tdm_load_chunk(unsigned long long gaddr, unsigned lds_off) {
    const unsigned units = CHUNK_BYTES / 8;   // 8192 8-byte units
    v4u g0; v8i g1; v4i g2, g3; v8i g4;
    g0[0] = 1u;                                       // count=1, user mode
    g0[1] = lds_off;                                  // lds_addr (bytes)
    g0[2] = (unsigned)(gaddr & 0xFFFFFFFFull);        // global_addr[31:0]
    g0[3] = (unsigned)((gaddr >> 32) & 0x1FFFFFFull)  // global_addr[56:32]
          | (2u << 30);                               // type = 2 (image)
    g1[0] = (int)(3u << 16);                          // data_size=3 (8B)
    g1[1] = (int)((units & 0xFFFFu) << 16);           // tensor_dim0[15:0]
    g1[2] = (int)((units >> 16) | (1u << 16));        // tensor_dim0[31:16], tensor_dim1=1
    g1[3] = (int)((units & 0xFFFFu) << 16);           // tile_dim0 = units
    g1[4] = 1;                                        // tile_dim1=1, tile_dim2=0
    g1[5] = (int)units;                               // tensor_dim0_stride[31:0]
    g1[6] = (int)((units & 0xFFFFu) << 16);           // stride hi=0, dim1_stride[15:0]
    g1[7] = 0;
    g2[0] = 1; g2[1] = 1; g2[2] = (int)units; g2[3] = 0;   // dim2=1, dim3=1, tile_dim3=0
    g3[0] = (int)units; g3[1] = 1 << 16; g3[2] = 0; g3[3] = 0; // dim3_stride, dim4=1, tile_dim4=0
    g4[0] = 0; g4[1] = 0; g4[2] = 0; g4[3] = 0;
    g4[4] = 0; g4[5] = 0; g4[6] = 0; g4[7] = 0;
    __builtin_amdgcn_tensor_load_to_lds(g0, g1, g2, g3, g4, 0);
}

// ---------------------------------------------------------------------------
// Main: 8 waves/block, each wave owns TWO 16-row tiles (shared B regs, two
// independent WMMA accumulator chains). B tiles staged global->LDS by the
// Tensor Data Mover, double buffered, shared by all 8 waves.
// ---------------------------------------------------------------------------
__global__ __launch_bounds__(256) void k_main(
    const float* __restrict__ z, const long long* __restrict__ bids,
    const float* __restrict__ cb, const float* __restrict__ csq,
    const unsigned short* __restrict__ B, unsigned long long Bgaddr,
    unsigned int* __restrict__ counts, float* __restrict__ lossAcc,
    float* __restrict__ out_q, float* __restrict__ out_idx) {

    extern __shared__ unsigned short smem[];   // 2 * BUF_U16
    const int lane = threadIdx.x & 31;
    const int wave = threadIdx.x >> 5;
    const int rt0  = (blockIdx.x * 8 + wave) * 2;   // two row tiles per wave
    const int row_base0 = rt0 * 16;
    const int row_base1 = row_base0 + 16;
    const int half = lane >> 4, m = lane & 15;
    const unsigned ldsbase = (unsigned)__builtin_amdgcn_groupstaticsize();

    // ---- A fragments (16-bit A 16x32 layout), split bf16 hi/lo, 2 row tiles
    union AU { v16bf v; unsigned short u[16]; };
    AU aHi[2][2], aLo[2][2];   // [rowtile][khalf]
#pragma unroll
    for (int rt = 0; rt < 2; ++rt) {
        const float* zrow = z + (size_t)((rt ? row_base1 : row_base0) + m) * DIM;
#pragma unroll
        for (int h = 0; h < 2; ++h) {
#pragma unroll
            for (int e = 0; e < 16; ++e) {
                int k = 32 * h + half * 16 + e;
                float x = zrow[k];
                unsigned short hi = f2bf(x);
                aHi[rt][h].u[e] = hi;
                aLo[rt][h].u[e] = f2bf(x - bf2f(hi));
            }
        }
    }

    float minv0[8], minv1[8];
    int   mini0[8], mini1[8];
#pragma unroll
    for (int r = 0; r < 8; ++r) {
        minv0[r] = 3.4e38f; mini0[r] = 0;
        minv1[r] = 3.4e38f; mini1[r] = 0;
    }

    // ---- prologue: issue TDM for chunks 0 and 1
    if (wave == 0) {
        tdm_load_chunk(Bgaddr,               ldsbase);
        tdm_load_chunk(Bgaddr + CHUNK_BYTES, ldsbase + CHUNK_BYTES);
    }

    for (int c = 0; c < NCHUNK; ++c) {
        if (wave == 0) {
            if (c + 1 < NCHUNK) __builtin_amdgcn_s_wait_tensorcnt(1);
            else                __builtin_amdgcn_s_wait_tensorcnt(0);
        }
        __syncthreads();                       // buf[c&1] filled

        const unsigned short* buf = smem + (c & 1) * BUF_U16;
#pragma unroll 2
        for (int tl = 0; tl < CHUNK; ++tl) {
            const unsigned short* tb = buf + tl * TILE_U16 + lane * 16;
            v16bf bh0 = *(const v16bf*)(tb);
            v16bf bh1 = *(const v16bf*)(tb + 512);
            v16bf bl0 = *(const v16bf*)(tb + 1024);
            v16bf bl1 = *(const v16bf*)(tb + 1536);

            v8f a0 = {0.f,0.f,0.f,0.f,0.f,0.f,0.f,0.f};
            v8f a1 = {0.f,0.f,0.f,0.f,0.f,0.f,0.f,0.f};
            // two independent accumulation chains -> hides XDL latency
            a0 = __builtin_amdgcn_wmma_f32_16x16x32_bf16(false, aHi[0][0].v, false, bh0, (short)0, a0, false, false);
            a1 = __builtin_amdgcn_wmma_f32_16x16x32_bf16(false, aHi[1][0].v, false, bh0, (short)0, a1, false, false);
            a0 = __builtin_amdgcn_wmma_f32_16x16x32_bf16(false, aHi[0][1].v, false, bh1, (short)0, a0, false, false);
            a1 = __builtin_amdgcn_wmma_f32_16x16x32_bf16(false, aHi[1][1].v, false, bh1, (short)0, a1, false, false);
            a0 = __builtin_amdgcn_wmma_f32_16x16x32_bf16(false, aLo[0][0].v, false, bh0, (short)0, a0, false, false);
            a1 = __builtin_amdgcn_wmma_f32_16x16x32_bf16(false, aLo[1][0].v, false, bh0, (short)0, a1, false, false);
            a0 = __builtin_amdgcn_wmma_f32_16x16x32_bf16(false, aLo[0][1].v, false, bh1, (short)0, a0, false, false);
            a1 = __builtin_amdgcn_wmma_f32_16x16x32_bf16(false, aLo[1][1].v, false, bh1, (short)0, a1, false, false);
            a0 = __builtin_amdgcn_wmma_f32_16x16x32_bf16(false, aHi[0][0].v, false, bl0, (short)0, a0, false, false);
            a1 = __builtin_amdgcn_wmma_f32_16x16x32_bf16(false, aHi[1][0].v, false, bl0, (short)0, a1, false, false);
            a0 = __builtin_amdgcn_wmma_f32_16x16x32_bf16(false, aHi[0][1].v, false, bl1, (short)0, a0, false, false);
            a1 = __builtin_amdgcn_wmma_f32_16x16x32_bf16(false, aHi[1][1].v, false, bl1, (short)0, a1, false, false);

            int   col = (c * CHUNK + tl) * 16 + m;
            float cs  = csq[col];
#pragma unroll
            for (int r = 0; r < 8; ++r) {
                float s0 = fmaf(-2.f, a0[r], cs);
                float s1 = fmaf(-2.f, a1[r], cs);
                if (s0 < minv0[r]) { minv0[r] = s0; mini0[r] = col; }
                if (s1 < minv1[r]) { minv1[r] = s1; mini1[r] = col; }
            }
        }
        __syncthreads();                       // buf[c&1] free
        if (wave == 0 && c + 2 < NCHUNK) {
            tdm_load_chunk(Bgaddr + (unsigned long long)(c + 2) * CHUNK_BYTES,
                           ldsbase + (c & 1) * CHUNK_BYTES);
        }
    }

    // ---- cross-lane argmin within 16-lane halves (masks 1..8 stay in-half)
#pragma unroll
    for (int r = 0; r < 8; ++r) {
#pragma unroll
        for (int s = 8; s >= 1; s >>= 1) {
            float ov0 = __shfl_xor(minv0[r], s, 32);
            int   oi0 = __shfl_xor(mini0[r], s, 32);
            if (ov0 < minv0[r] || (ov0 == minv0[r] && oi0 < mini0[r])) { minv0[r] = ov0; mini0[r] = oi0; }
            float ov1 = __shfl_xor(minv1[r], s, 32);
            int   oi1 = __shfl_xor(mini1[r], s, 32);
            if (ov1 < minv1[r] || (ov1 == minv1[r] && oi1 < mini1[r])) { minv1[r] = ov1; mini1[r] = oi1; }
        }
    }

    // ---- gather / outputs / losses / histogram (32 rows per wave)
    float lsum = 0.f;
#pragma unroll
    for (int i = 0; i < 32; ++i) {
        int idx = (i < 16) ? __shfl(mini0[i & 7], (i >> 3 & 1) << 4, 32)
                           : __shfl(mini1[i & 7], (i >> 3 & 1) << 4, 32);
        int row = row_base0 + i;
        const float* crow = cb + (size_t)idx * DIM;
        const float* zrow = z + (size_t)row * DIM;
        float q0 = crow[lane], q1 = crow[lane + 32];
        float z0 = zrow[lane], z1 = zrow[lane + 32];
        out_q[(size_t)row * DIM + lane]      = q0;
        out_q[(size_t)row * DIM + lane + 32] = q1;
        float d0 = z0 - q0, d1 = z1 - q1;
        lsum = fmaf(d0, d0, lsum);
        lsum = fmaf(d1, d1, lsum);
        if (lane == 0) {
            out_idx[row] = (float)idx;
            int b = (int)bids[row];
            atomicAdd(&counts[b * KCODES + idx], 1u);
        }
    }
#pragma unroll
    for (int s = 16; s >= 1; s >>= 1) lsum += __shfl_xor(lsum, s, 32);
    if (lane == 0) atomicAdd(lossAcc, lsum);
}

// ---------------------------------------------------------------------------
__global__ void k_final(const unsigned int* __restrict__ counts,
                        const float* __restrict__ lossAcc,
                        float* __restrict__ out) {
    __shared__ float sEnt[NBATCH], sPerp[NBATCH], sUni[NBATCH];
    int lane = threadIdx.x & 31, wave = threadIdx.x >> 5;
    const unsigned int* c = counts + wave * KCODES;

    float tot = 0.f;
    for (int i = lane; i < KCODES; i += 32) tot += (float)c[i];
#pragma unroll
    for (int s = 16; s >= 1; s >>= 1) tot += __shfl_xor(tot, s, 32);
    float tmax = fmaxf(tot, 1.f);

    float ent = 0.f, uni = 0.f;
    for (int i = lane; i < KCODES; i += 32) {
        float cc = (float)c[i];
        float p  = cc / tmax;
        ent -= p * logf(p + 1e-10f);
        uni += (cc > 0.f) ? 1.f : 0.f;
    }
#pragma unroll
    for (int s = 16; s >= 1; s >>= 1) {
        ent += __shfl_xor(ent, s, 32);
        uni += __shfl_xor(uni, s, 32);
    }
    if (lane == 0) { sEnt[wave] = ent; sPerp[wave] = expf(ent); sUni[wave] = uni; }
    __syncthreads();
    if (threadIdx.x == 0) {
        float ae = 0.f, ap = 0.f, au = 0.f;
        for (int b = 0; b < NBATCH; ++b) { ae += sEnt[b]; ap += sPerp[b]; au += sUni[b]; }
        ae *= (1.f / NBATCH); ap *= (1.f / NBATCH); au *= (1.f / NBATCH);
        float L = lossAcc[0] / (float)((size_t)N_PTS * DIM);
        out[OUT_VQ]   = L;
        out[OUT_COMM] = L;
        out[OUT_PERP] = ap;
        out[OUT_ENT]  = ae;
        out[OUT_UNIQ] = au;
    }
}

// ---------------------------------------------------------------------------
extern "C" void kernel_launch(void* const* d_in, const int* in_sizes, int n_in,
                              void* d_out, int out_size, void* d_ws, size_t ws_size,
                              hipStream_t stream) {
    (void)in_sizes; (void)n_in; (void)out_size; (void)ws_size;
    const float*     z    = (const float*)d_in[0];
    const long long* bids = (const long long*)d_in[1];
    const float*     cb   = (const float*)d_in[2];

    char* ws = (char*)d_ws;
    float*          csq     = (float*)(ws + OFF_CSQ);
    unsigned short* B       = (unsigned short*)(ws + OFF_B);
    unsigned int*   counts  = (unsigned int*)(ws + OFF_CNT);
    float*          lossAcc = (float*)(ws + OFF_LOSS);

    float* out     = (float*)d_out;
    float* out_q   = out;
    float* out_idx = out + OUT_IDX;

    (void)hipMemsetAsync(counts, 0, (size_t)NBATCH * KCODES * 4 + 4, stream);

    k_csq  <<<KCODES / 256, 256, 0, stream>>>(cb, csq);
    k_prepB<<<(CTILES * 4 * 32) / 256, 256, 0, stream>>>(cb, B);

    (void)hipFuncSetAttribute(reinterpret_cast<const void*>(k_main),
                              hipFuncAttributeMaxDynamicSharedMemorySize, 2 * CHUNK_BYTES);
    k_main <<<N_PTS / 16 / 16, 256, 2 * CHUNK_BYTES, stream>>>(
        z, bids, cb, csq, B, (unsigned long long)(uintptr_t)B,
        counts, lossAcc, out_q, out_idx);

    k_final<<<1, 256, 0, stream>>>(counts, lossAcc, out);
}
